// RDOVAE_73426760893105
// MI455X (gfx1250) — matrix operand order repaired and checked by reading, not compiled, not run on hardware
//
#include <hip/hip_runtime.h>
#include <hip/hip_bf16.h>
#include <math.h>

typedef _Float16 f16;
typedef __attribute__((ext_vector_type(16))) _Float16 v16h;
typedef __attribute__((ext_vector_type(8)))  _Float16 v8h;
typedef __attribute__((ext_vector_type(8)))  float    v8f;

#define NROWS 25600   // B*SE
#define SE_   400
#define B_    64
#define GR_BB 16      // batch rows per GRU recurrence block

// ---------------------------------------------------------------------------
// CDNA5 async global->LDS copy helpers (ASYNCcnt-tracked DMA, ISA ch.10.7)
// ---------------------------------------------------------------------------
__device__ __forceinline__ void async_copy16(void* lds, const void* g) {
    unsigned lo = (unsigned)(unsigned long long)lds;   // low 32b of generic LDS addr == LDS offset
    asm volatile("global_load_async_to_lds_b128 %0, %1, off"
                 :: "v"(lo), "v"((unsigned long long)g) : "memory");
}
__device__ __forceinline__ void wait_async() {
    asm volatile("s_wait_asynccnt 0x0" ::: "memory");
}

// ---------------------------------------------------------------------------
// Weight conversion: fp32 (M x K) -> f16 (M x Kp), zero padded K..Kp-1
// ---------------------------------------------------------------------------
__global__ void cvt_w(const float* __restrict__ W, f16* __restrict__ O,
                      int M, int K, int Kp) {
    int i = blockIdx.x * 256 + threadIdx.x;
    int total = M * Kp;
    if (i < total) {
        int m = i / Kp, k = i % Kp;
        O[i] = (k < K) ? (f16)W[(size_t)m * K + k] : (f16)0.f;
    }
}

// conv_w (80, 2048, 4) OIH -> 4 contiguous (80 x 2048) f16 matrices, one per tap
__global__ void cvt_conv(const float* __restrict__ W, f16* __restrict__ O) {
    int i = blockIdx.x * 256 + threadIdx.x;
    if (i < 4 * 80 * 2048) {
        int j = i / (80 * 2048);
        int rem = i % (80 * 2048);
        int o = rem / 2048, c = rem % 2048;
        O[i] = (f16)W[((size_t)o * 2048 + c) * 4 + j];
    }
}

// ---------------------------------------------------------------------------
// Build encoder input Xe (N x 544) = [feat pair(40) | emb(480) | lambda | pad]
// Pre-fill decoder input Xd (N x 608) emb cols [80..559] and zero tail pad.
// quant_scale qs (N x 80) = softplus(emb[:, :80]).
// ---------------------------------------------------------------------------
__global__ void embed_build(const float* __restrict__ embt,
                            const int* __restrict__ q_id,
                            const float* __restrict__ lam,
                            const float* __restrict__ feats,
                            f16* __restrict__ Xe, f16* __restrict__ Xd,
                            float* __restrict__ qs) {
    int r = blockIdx.x;               // 0..N-1 == b*SE + s
    int b = r / SE_, s = r % SE_;
    int q = q_id[b * SE_ + s];
    const float* e = embt + (size_t)q * 480;
    for (int i = threadIdx.x; i < 480; i += blockDim.x) {
        float v = e[i];
        Xe[(size_t)r * 544 + 40 + i] = (f16)v;
        Xd[(size_t)r * 608 + 80 + i] = (f16)v;
        if (i < 80) {
            float sp = (v > 20.f) ? v : log1pf(expf(v));
            qs[(size_t)r * 80 + i] = sp;
        }
    }
    for (int i = threadIdx.x; i < 40; i += blockDim.x)
        Xe[(size_t)r * 544 + i] = (f16)feats[((size_t)b * 800 + 2 * s) * 20 + i];
    if (threadIdx.x == 0) {
        Xe[(size_t)r * 544 + 520] = (f16)lam[b * SE_ + s];
        for (int i = 521; i < 544; ++i) Xe[(size_t)r * 544 + i] = (f16)0.f;
        for (int i = 584; i < 608; ++i) Xd[(size_t)r * 608 + i] = (f16)0.f;
    }
}

// ---------------------------------------------------------------------------
// Generic WMMA GEMM: C[n,m] = act( sum_k A[n,k]*Bw[m,k] + bias[m] )
// One block per 16-row A strip. The strip is DMA'd to LDS once (async copy),
// then 4 waves loop over all M col-tiles, A-fragments fed from LDS.
// K must be a multiple of 32 (zero-padded weights/activations).
// Dynamic LDS: 32*K bytes.
// ---------------------------------------------------------------------------
__global__ __launch_bounds__(128) void gemm_wmma(
    const f16* __restrict__ A, int lda,
    const f16* __restrict__ Bw, int K,
    const float* __restrict__ bias,
    f16* __restrict__ Ch, int ldch,
    float* __restrict__ Cf, int ldcf,
    int M, int act) {
    extern __shared__ char smem_raw[];
    f16* As = (f16*)smem_raw;                  // 16 x K halves
    const int tid  = threadIdx.x;
    const int lane = tid & 31;
    const int wv   = tid >> 5;
    const int tn   = blockIdx.x;

    // stage 16-row A strip into LDS with async DMA
    const int cpr = K >> 3;                    // 16B chunks per row
    const f16* Abase = A + (size_t)(tn * 16) * lda;
    for (int c = tid; c < 16 * cpr; c += 128) {
        int row = c / cpr, cc = c - row * cpr;
        async_copy16(As + row * K + cc * 8, Abase + (size_t)row * lda + cc * 8);
    }
    wait_async();
    __syncthreads();

    const int kA0 = (lane >> 4) * 8;           // A frag K sub-block per lane half
    const int kB0 = (lane >> 4) * 16;          // unused; B frag uses kA0 like A
    const f16* Arow = As + (lane & 15) * K;    // LDS
    const int crow0 = tn * 16 + (lane >> 4) * 8;

    for (int tm = wv; tm * 16 < M; tm += 4) {
        int colB  = tm * 16 + (lane & 15);
        int colBc = colB < M ? colB : M - 1;
        const f16* Brow = Bw + (size_t)colBc * K;
        v8f acc = {};
        for (int k0 = 0; k0 < K; k0 += 32) {
            union { v16h v; v8h h[2]; } a, b;
            a.h[0] = *(const v8h*)(Arow + k0 + kA0);
            a.h[1] = *(const v8h*)(Arow + k0 + 16 + kA0);
            b.h[0] = *(const v8h*)(Brow + k0 + kA0);
            b.h[1] = *(const v8h*)(Brow + k0 + 16 + kA0);
            acc = __builtin_amdgcn_wmma_f32_16x16x32_f16(false, a.v, false, b.v,
                                                         (short)0, acc, false, false);
        }
        if (colB < M) {
            float bv = bias ? bias[colB] : 0.f;
#pragma unroll
            for (int r = 0; r < 8; ++r) {
                float v = acc[r] + bv;
                if (act == 1) v = tanhf(v);
                if (Ch) Ch[(size_t)(crow0 + r) * ldch + colB] = (f16)v;
                if (Cf) Cf[(size_t)(crow0 + r) * ldcf + colB] = v;
            }
        }
    }
}

// ---------------------------------------------------------------------------
// Causal conv as 4 accumulated WMMA GEMMs with per-row time masking.
// A = x9 (N x 2048), Wc = 4 taps of (80 x 2048) f16, out f32 z (N x 80) + bias
// ---------------------------------------------------------------------------
__global__ __launch_bounds__(128) void conv_gemm_wmma(
    const f16* __restrict__ A, const f16* __restrict__ Wc,
    const float* __restrict__ bias, float* __restrict__ Cf) {
    const int K = 2048, M = 80, lda = 2048;
    int lane = threadIdx.x & 31;
    int wv   = threadIdx.x >> 5;
    int tm   = blockIdx.y * 4 + wv;
    if (tm * 16 >= M) return;
    int tn   = blockIdx.x;
    int rowA = tn * 16 + (lane & 15);
    int t    = rowA % SE_;
    int colB = tm * 16 + (lane & 15);
    int kA0 = (lane >> 4) * 8;
    int kB0 = (lane >> 4) * 16;
    v8f acc = {};
    for (int j = 0; j < 4; ++j) {
        int shift = j - 3;
        bool valid = (t + shift) >= 0;
        const f16* Arow = A + (size_t)(valid ? (rowA + shift) : rowA) * lda;
        const f16* Brow = Wc + (size_t)(j * M + colB) * K;
        for (int k0 = 0; k0 < K; k0 += 32) {
            union { v16h v; v8h h[2]; } a, b;
            if (valid) {
                a.h[0] = *(const v8h*)(Arow + k0 + kA0);
                a.h[1] = *(const v8h*)(Arow + k0 + 16 + kA0);
            } else {
                v8h z = {};
                a.h[0] = z; a.h[1] = z;
            }
            b.h[0] = *(const v8h*)(Brow + k0 + kA0);
            b.h[1] = *(const v8h*)(Brow + k0 + 16 + kA0);
            acc = __builtin_amdgcn_wmma_f32_16x16x32_f16(false, a.v, false, b.v,
                                                         (short)0, acc, false, false);
        }
    }
    if (colB < M) {
        float bv = bias[colB];
        int crow0 = tn * 16 + (lane >> 4) * 8;
#pragma unroll
        for (int r = 0; r < 8; ++r)
            Cf[(size_t)(crow0 + r) * 80 + colB] = acc[r] + bv;
    }
}

// ---------------------------------------------------------------------------
// GRU recurrence: one block = 16 batch rows, 16 waves. Per step:
//   async-prefetch next step's xp slab into LDS (double buffered, overlaps
//   with WMMA), gh = h*Whh^T (WMMA, A from LDS), gate math, h update.
// Dynamic LDS: hsm 16x256 f16 | ghs 16x768 f32 | xps 2 x 16x768 f16
// ---------------------------------------------------------------------------
#define GRU_SHM (GR_BB*256*2 + GR_BB*768*4 + 2*GR_BB*768*2)
__global__ __launch_bounds__(512) void gru_rec(
    const f16* __restrict__ xp, const f16* __restrict__ Whh,
    const float* __restrict__ bhh, f16* __restrict__ ys, int ldy) {
    extern __shared__ char smem_raw[];
    f16*   hsm = (f16*)smem_raw;                                  // 16x256
    float* ghs = (float*)(smem_raw + GR_BB * 256 * 2);            // 16x768
    f16*   xps = (f16*)(smem_raw + GR_BB * 256 * 2 + GR_BB * 768 * 4); // 2 x 16x768
    const int tid  = threadIdx.x;
    const int lane = tid & 31;
    const int wv   = tid >> 5;            // 16 waves
    const int bg   = blockIdx.x;

    for (int i = tid; i < GR_BB * 256; i += 512) hsm[i] = (f16)0.f;

    // prefetch xp slab for t=0 (16 rows x 768 halves = 1536 16B chunks)
    for (int c = tid; c < GR_BB * 96; c += 512) {
        int row = c / 96, cc = c - row * 96;
        const f16* g = xp + ((size_t)(bg * GR_BB + row) * SE_ + 0) * 768 + cc * 8;
        async_copy16(xps + row * 768 + cc * 8, g);
    }
    wait_async();
    __syncthreads();

    const int mrow = lane & 15;
    const int kA0 = (lane >> 4) * 8;
    const int kB0 = (lane >> 4) * 16;

    for (int t = 0; t < SE_; ++t) {
        // issue async prefetch of next step's xp slab into the other buffer
        if (t + 1 < SE_) {
            f16* dst = xps + ((t + 1) & 1) * GR_BB * 768;
            for (int c = tid; c < GR_BB * 96; c += 512) {
                int row = c / 96, cc = c - row * 96;
                const f16* g = xp + ((size_t)(bg * GR_BB + row) * SE_ + (t + 1)) * 768 + cc * 8;
                async_copy16(dst + row * 768 + cc * 8, g);
            }
        }
        // gh = h * Whh^T : 48 col-tiles, 3 per wave
#pragma unroll
        for (int tt = 0; tt < 3; ++tt) {
            int tm   = wv * 3 + tt;
            int colB = tm * 16 + (lane & 15);
            const f16* Brow = Whh + (size_t)colB * 256;
            const f16* Arow = hsm + mrow * 256;
            v8f acc = {};
#pragma unroll
            for (int k0 = 0; k0 < 256; k0 += 32) {
                union { v16h v; v8h h[2]; } a, b;
                a.h[0] = *(const v8h*)(Arow + k0 + kA0);
                a.h[1] = *(const v8h*)(Arow + k0 + 16 + kA0);
                b.h[0] = *(const v8h*)(Brow + k0 + kA0);
                b.h[1] = *(const v8h*)(Brow + k0 + 16 + kA0);
                acc = __builtin_amdgcn_wmma_f32_16x16x32_f16(false, a.v, false, b.v,
                                                             (short)0, acc, false, false);
            }
            int crow0 = (lane >> 4) * 8;
#pragma unroll
            for (int r = 0; r < 8; ++r)
                ghs[(crow0 + r) * 768 + colB] = acc[r];
        }
        wait_async();          // xp slab (t+1) landed; also fence this step's use
        __syncthreads();       // ghs visible to all waves

        const f16* xcur = xps + (t & 1) * GR_BB * 768;
        for (int u = tid; u < GR_BB * 256; u += 512) {
            int i = u >> 8, j = u & 255;
            float xr = (float)xcur[i * 768 + j];
            float xz = (float)xcur[i * 768 + 256 + j];
            float xn = (float)xcur[i * 768 + 512 + j];
            float hr = ghs[i * 768 + j]       + bhh[j];
            float hz = ghs[i * 768 + 256 + j] + bhh[256 + j];
            float hn = ghs[i * 768 + 512 + j] + bhh[512 + j];
            float hp = (float)hsm[i * 256 + j];
            float rg = 1.f / (1.f + expf(-(xr + hr)));
            float zg = 1.f / (1.f + expf(-(xz + hz)));
            float ng = tanhf(xn + rg * hn);
            float h2 = (1.f - zg) * ng + zg * hp;
            hsm[i * 256 + j] = (f16)h2;
            ys[((size_t)(bg * GR_BB + i) * SE_ + t) * ldy + j] = (f16)h2;
        }
        __syncthreads();
    }
}

// ---------------------------------------------------------------------------
// z -> quantize -> decoder latent columns of Xd
// ---------------------------------------------------------------------------
__global__ void zpost(const float* __restrict__ zr, const float* __restrict__ qs,
                      f16* __restrict__ Xd) {
    int i = blockIdx.x * 256 + threadIdx.x;
    if (i < NROWS * 80) {
        int r = i / 80, o = i % 80;
        float z  = zr[i] * qs[i];
        float zq = rintf(z);                     // jnp.round = half-to-even
        Xd[(size_t)r * 608 + o] = (f16)(zq / qs[i]);
    }
}

// gather x6 (block 5 of Xbuf) at t = SE-1 -> S6 (64 x 256 f16)
__global__ void gather_last(const f16* __restrict__ Xbuf, f16* __restrict__ S6) {
    int i = blockIdx.x * 256 + threadIdx.x;
    if (i < 64 * 256) {
        int b = i / 256, j = i % 256;
        S6[i] = Xbuf[((size_t)b * SE_ + (SE_ - 1)) * 2048 + 5 * 256 + j];
    }
}

// soft PVQ forward: init = round(k*x/||x||1) / ||round(...)||2
// (faithful: the torch-style loop only rescales `scale`; x_quant is fixed
//  after the first round, so it is dead for the forward output)
__global__ void pvq_kernel(const float* __restrict__ st, float* __restrict__ initv) {
    int b = threadIdx.x;
    if (b < 64) {
        float x[24]; float l1 = 0.f;
        for (int i = 0; i < 24; ++i) { x[i] = st[b * 24 + i]; l1 += fabsf(x[i]); }
        float inv = 1.f / (l1 + 1e-30f);
        float q[24]; float n2 = 0.f;
        for (int i = 0; i < 24; ++i) { q[i] = rintf(82.f * x[i] * inv); n2 += q[i] * q[i]; }
        float invn = 1.f / (1e-15f + sqrtf(n2));
        for (int i = 0; i < 24; ++i) initv[b * 24 + i] = q[i] * invn;
    }
}

// broadcast init across time into Xd cols [560..583]
__global__ void bcast_init(const float* __restrict__ initv, f16* __restrict__ Xd) {
    int i = blockIdx.x * 256 + threadIdx.x;
    if (i < NROWS * 24) {
        int r = i / 24, k = i % 24;
        int b = r / SE_;
        Xd[(size_t)r * 608 + 560 + k] = (f16)initv[b * 24 + k];
    }
}

// ---------------------------------------------------------------------------
// Host orchestration
// ---------------------------------------------------------------------------
struct WCvt { int idx; int M; int K; int Kp; };
enum { W_E_D1, W_E_G1_IH, W_E_G1_HH, W_E_D2, W_E_G2_IH, W_E_G2_HH, W_E_D3,
       W_E_G3_IH, W_E_G3_HH, W_E_D4, W_E_D5, W_SD1, W_SD2,
       W_D_D1, W_D_D2, W_D_D3, W_D_G1_IH, W_D_G1_HH, W_D_G2_IH, W_D_G2_HH,
       W_D_G3_IH, W_D_G3_HH, W_D_D4, W_D_D5, W_OUT, NW };
static const WCvt wtab[NW] = {
    {4, 256, 521, 544}, {6, 768, 256, 256}, {7, 768, 256, 256},
    {10, 256, 256, 256}, {12, 768, 256, 256}, {13, 768, 256, 256},
    {16, 256, 256, 256}, {18, 768, 256, 256}, {19, 768, 256, 256},
    {22, 256, 256, 256}, {24, 256, 256, 256}, {28, 128, 256, 256},
    {30, 24, 128, 128},  {32, 256, 584, 608}, {34, 256, 256, 256},
    {36, 256, 256, 256}, {38, 768, 256, 256}, {39, 768, 256, 256},
    {42, 768, 256, 256}, {43, 768, 256, 256}, {46, 768, 256, 256},
    {47, 768, 256, 256}, {50, 256, 256, 256}, {52, 256, 256, 256},
    {54, 80, 2048, 2048}
};

extern "C" void kernel_launch(void* const* d_in, const int* in_sizes, int n_in,
                              void* d_out, int out_size, void* d_ws, size_t ws_size,
                              hipStream_t stream) {
    (void)in_sizes; (void)n_in; (void)out_size; (void)ws_size;
    const float* features = (const float*)d_in[0];
    const int*   q_id     = (const int*)d_in[1];
    const float* lam      = (const float*)d_in[2];
    const float* embt     = (const float*)d_in[3];
    auto Bf = [&](int i) { return (const float*)d_in[i]; };

    // --- workspace bump allocator ---
    char* p = (char*)d_ws;
    auto alloc = [&](size_t bytes) -> void* {
        void* r = (void*)p;
        p += (bytes + 255) & ~(size_t)255;
        return r;
    };
    f16*   Xe   = (f16*)alloc((size_t)NROWS * 544 * 2);
    f16*   Xbuf = (f16*)alloc((size_t)NROWS * 2048 * 2);   // x1..x8 column blocks
    f16*   Ybuf = (f16*)alloc((size_t)NROWS * 2048 * 2);   // y1..y8 column blocks
    f16*   Xd   = (f16*)alloc((size_t)NROWS * 608 * 2);
    float* qs   = (float*)alloc((size_t)NROWS * 80 * 4);
    f16*   xp   = (f16*)alloc((size_t)NROWS * 768 * 2);    // reused for all 6 GRUs
    float* zr   = (float*)alloc((size_t)NROWS * 80 * 4);
    f16*   S6   = (f16*)alloc((size_t)64 * 256 * 2);
    f16*   H1   = (f16*)alloc((size_t)64 * 128 * 2);
    float* stF  = (float*)alloc((size_t)64 * 24 * 4);
    float* inF  = (float*)alloc((size_t)64 * 24 * 4);
    f16* wf[NW];
    for (int i = 0; i < NW; ++i)
        wf[i] = (f16*)alloc((size_t)wtab[i].M * wtab[i].Kp * 2);
    f16* Wcv = (f16*)alloc((size_t)4 * 80 * 2048 * 2);

    // --- weight conversion ---
    for (int i = 0; i < NW; ++i) {
        int total = wtab[i].M * wtab[i].Kp;
        cvt_w<<<(total + 255) / 256, 256, 0, stream>>>(
            (const float*)d_in[wtab[i].idx], wf[i], wtab[i].M, wtab[i].K, wtab[i].Kp);
    }
    cvt_conv<<<(4 * 80 * 2048 + 255) / 256, 256, 0, stream>>>((const float*)d_in[26], Wcv);

    // --- inputs / embeddings ---
    embed_build<<<NROWS, 64, 0, stream>>>(embt, q_id, lam, features, Xe, Xd, qs);

    auto gemm = [&](const f16* A, int lda, int widx, int K, const float* bias,
                    f16* Ch, int ldch, float* Cf, int ldcf, int Nr, int M, int act) {
        size_t shm = (size_t)32 * K;               // 16 rows x K halves
        gemm_wmma<<<Nr / 16, 128, shm, stream>>>(A, lda, wf[widx], K, bias,
                                                 Ch, ldch, Cf, ldcf, M, act);
    };
    auto gru = [&](const f16* xsrc, int ldx, int wih, int whh, int bih, int bhh,
                   f16* ydst, int ldy) {
        gemm(xsrc, ldx, wih, 256, Bf(bih), xp, 768, nullptr, 0, NROWS, 768, 0);
        gru_rec<<<B_ / GR_BB, 512, GRU_SHM, stream>>>(xp, wf[whh], Bf(bhh), ydst, ldy);
    };

    // --- encoder ---
    gemm(Xe, 544, W_E_D1, 544, Bf(5), Xbuf + 0 * 256, 2048, nullptr, 0, NROWS, 256, 1);
    gru(Xbuf + 0 * 256, 2048, W_E_G1_IH, W_E_G1_HH, 8, 9,  Xbuf + 1 * 256, 2048);
    gemm(Xbuf + 1 * 256, 2048, W_E_D2, 256, Bf(11), Xbuf + 2 * 256, 2048, nullptr, 0, NROWS, 256, 1);
    gru(Xbuf + 2 * 256, 2048, W_E_G2_IH, W_E_G2_HH, 14, 15, Xbuf + 3 * 256, 2048);
    gemm(Xbuf + 3 * 256, 2048, W_E_D3, 256, Bf(17), Xbuf + 4 * 256, 2048, nullptr, 0, NROWS, 256, 1);
    gru(Xbuf + 4 * 256, 2048, W_E_G3_IH, W_E_G3_HH, 20, 21, Xbuf + 5 * 256, 2048);
    gemm(Xbuf + 5 * 256, 2048, W_E_D4, 256, Bf(23), Xbuf + 6 * 256, 2048, nullptr, 0, NROWS, 256, 1);
    gemm(Xbuf + 6 * 256, 2048, W_E_D5, 256, Bf(25), Xbuf + 7 * 256, 2048, nullptr, 0, NROWS, 256, 1);

    // --- conv head (latents, raw) ---
    {
        dim3 grid(NROWS / 16, 2);
        conv_gemm_wmma<<<grid, 128, 0, stream>>>(Xbuf, Wcv, Bf(27), zr);
    }

    // --- state head + PVQ ---
    gather_last<<<(64 * 256 + 255) / 256, 256, 0, stream>>>(Xbuf, S6);
    gemm(S6, 256, W_SD1, 256, Bf(29), H1, 128, nullptr, 0, 64, 128, 1);
    gemm(H1, 128, W_SD2, 128, Bf(31), nullptr, 0, stF, 24, 64, 24, 1);
    pvq_kernel<<<1, 64, 0, stream>>>(stF, inF);
    bcast_init<<<(NROWS * 24 + 255) / 256, 256, 0, stream>>>(inF, Xd);

    // --- quantize latents into decoder input ---
    zpost<<<(NROWS * 80 + 255) / 256, 256, 0, stream>>>(zr, qs, Xd);

    // --- decoder ---
    gemm(Xd, 608, W_D_D1, 608, Bf(33), Ybuf + 0 * 256, 2048, nullptr, 0, NROWS, 256, 1);
    gemm(Ybuf + 0 * 256, 2048, W_D_D2, 256, Bf(35), Ybuf + 1 * 256, 2048, nullptr, 0, NROWS, 256, 1);
    gemm(Ybuf + 1 * 256, 2048, W_D_D3, 256, Bf(37), Ybuf + 2 * 256, 2048, nullptr, 0, NROWS, 256, 1);
    gru(Ybuf + 2 * 256, 2048, W_D_G1_IH, W_D_G1_HH, 40, 41, Ybuf + 3 * 256, 2048);
    gru(Ybuf + 3 * 256, 2048, W_D_G2_IH, W_D_G2_HH, 44, 45, Ybuf + 4 * 256, 2048);
    gru(Ybuf + 4 * 256, 2048, W_D_G3_IH, W_D_G3_HH, 48, 49, Ybuf + 5 * 256, 2048);
    gemm(Ybuf + 5 * 256, 2048, W_D_D4, 256, Bf(51), Ybuf + 6 * 256, 2048, nullptr, 0, NROWS, 256, 1);
    gemm(Ybuf + 6 * 256, 2048, W_D_D5, 256, Bf(53), Ybuf + 7 * 256, 2048, nullptr, 0, NROWS, 256, 1);

    // --- output head: tanh GEMM straight to d_out (reshape is a view) ---
    gemm(Ybuf, 2048, W_OUT, 2048, Bf(55), nullptr, 0, (float*)d_out, 80, NROWS, 80, 1);
}
